// ScaledDotProduct_64269890617540
// MI455X (gfx1250) — matrix operand order repaired and checked by reading, not compiled, not run on hardware
//
#include <hip/hip_runtime.h>
#include <hip/hip_bf16.h>

// ---------------------------------------------------------------------------
// ScaledDotProduct scores kernel for MI455X (gfx1250, wave32, WMMA).
//
// scores = q @ k^T / sqrt(64) + attn_mask ; -inf on padded keys.
//
// Roofline: 512 MiB of f32 output writes dominate (~23 us @ 23.3 TB/s HBM);
// q/k/mask (~50 MiB) live in the 192 MB L2. 17.2 GFLOP of matmul is cheap on
// the WMMA pipes, so we spend compute on accuracy: bf16x3 split-precision
// (hi/lo) V_WMMA_F32_16X16X32_BF16 with f32 accumulation (~f32-accurate).
// ---------------------------------------------------------------------------

typedef __attribute__((ext_vector_type(16))) __bf16 v16bf;
typedef __attribute__((ext_vector_type(8)))  float  v8f;
typedef __attribute__((ext_vector_type(4)))  float  v4f;

#define B_  2
#define H_  16
#define SQ_ 2048
#define SK_ 2048
#define D_  64

// A fragment (16x32 bf16, ISA 7.12.2): lane m=lane&15 is row M; lg=lane>>4.
//   elems 0..7  <- K = K0 + lg*8 + j
//   elems 8..15 <- K = K0 + 16 + lg*8 + j
__device__ __forceinline__ void build_a(const float* __restrict__ rowp, int K0,
                                        int lg, v16bf& hi, v16bf& lo) {
  const float* p0 = rowp + K0 + lg * 8;
  v4f x0 = *(const v4f*)(p0);
  v4f x1 = *(const v4f*)(p0 + 4);
  v4f x2 = *(const v4f*)(p0 + 16);
  v4f x3 = *(const v4f*)(p0 + 20);
#pragma unroll
  for (int j = 0; j < 4; ++j) {
    __bf16 h0 = (__bf16)x0[j];
    __bf16 h1 = (__bf16)x1[j];
    __bf16 h2 = (__bf16)x2[j];
    __bf16 h3 = (__bf16)x3[j];
    hi[j]      = h0;  lo[j]      = (__bf16)(x0[j] - (float)h0);
    hi[4 + j]  = h1;  lo[4 + j]  = (__bf16)(x1[j] - (float)h1);
    hi[8 + j]  = h2;  lo[8 + j]  = (__bf16)(x2[j] - (float)h2);
    hi[12 + j] = h3;  lo[12 + j] = (__bf16)(x3[j] - (float)h3);
  }
}

// B fragment (32x16 bf16): lane n=lane&15 is column N (= row of k); lg=lane>>4.
//   elems 0..7  <- K = K0 + lg*8 + j
//   elems 8..15 <- K = K0 + 16 + lg*8 + j
__device__ __forceinline__ void build_b(const float* __restrict__ colp, int K0,
                                        int lg, v16bf& hi, v16bf& lo) {
  const float* p = colp + K0 + lg * 8;
  v4f x0 = *(const v4f*)(p);
  v4f x1 = *(const v4f*)(p + 4);
  v4f x2 = *(const v4f*)(p + 16);
  v4f x3 = *(const v4f*)(p + 20);
#pragma unroll
  for (int j = 0; j < 4; ++j) {
    __bf16 h0 = (__bf16)x0[j];
    __bf16 h1 = (__bf16)x1[j];
    __bf16 h2 = (__bf16)x2[j];
    __bf16 h3 = (__bf16)x3[j];
    hi[j]      = h0;  lo[j]      = (__bf16)(x0[j] - (float)h0);
    hi[4 + j]  = h1;  lo[4 + j]  = (__bf16)(x1[j] - (float)h1);
    hi[8 + j]  = h2;  lo[8 + j]  = (__bf16)(x2[j] - (float)h2);
    hi[12 + j] = h3;  lo[12 + j] = (__bf16)(x3[j] - (float)h3);
  }
}

#define WMMA_BF16(A, Bm, C) \
  __builtin_amdgcn_wmma_f32_16x16x32_bf16(false, (A), false, (Bm), (short)0, \
                                          (C), false, false)

// Block = 256 threads = 8 waves. Each wave: one 16-row q tile x 128 k columns
// (8 WMMA tiles). Waves in a block share the q tile (L0/L2 reuse).
// grid = (SK/(128*8)=2, SQ/16=128, B*H=32)
__global__ __launch_bounds__(256)
void sdp_scores_kernel(const float* __restrict__ q,
                       const float* __restrict__ k,
                       const float* __restrict__ mask,
                       const unsigned char* __restrict__ kp,
                       float* __restrict__ out) {
  const int lane = threadIdx.x & 31;
  const int wave = threadIdx.x >> 5;
  const int lg   = lane >> 4;   // half-wave group
  const int ln   = lane & 15;

  const int colgroup = blockIdx.x * 8 + wave;  // 0..15  (128 cols each)
  const int qtile    = blockIdx.y;             // 0..127
  const int bh       = blockIdx.z;             // 0..31
  const int b        = bh >> 4;                // bh / H_

  // ---- A fragments (q tile, built once, reused for 8 column tiles) ----
  const int qrow = qtile * 16 + ln;
  const float* qrowp = q + ((size_t)bh * SQ_ + qrow) * D_;
  v16bf ah0, al0, ah1, al1;
  build_a(qrowp, 0,  lg, ah0, al0);
  build_a(qrowp, 32, lg, ah1, al1);

  const float NEG_INF = -__builtin_inff();
  const int colbase0 = colgroup * 128;

#pragma unroll 1
  for (int t = 0; t < 8; ++t) {
    const int col = colbase0 + t * 16 + ln;
    const float* kcolp = k + ((size_t)bh * SK_ + col) * D_;

    v16bf bhi0, blo0, bhi1, blo1;
    build_b(kcolp, 0,  lg, bhi0, blo0);
    build_b(kcolp, 32, lg, bhi1, blo1);

    // bf16x3: hi*hi + hi*lo + lo*hi, f32 accumulate. 6 WMMAs per tile.
    v8f acc = {};
    acc = WMMA_BF16(ah0, bhi0, acc);
    acc = WMMA_BF16(ah1, bhi1, acc);
    acc = WMMA_BF16(ah0, blo0, acc);
    acc = WMMA_BF16(ah1, blo1, acc);
    acc = WMMA_BF16(al0, bhi0, acc);
    acc = WMMA_BF16(al1, bhi1, acc);

    // ---- epilogue: scale, +attn_mask, key-padding -inf, store ----
    const unsigned char pad = kp[(size_t)b * SK_ + col];
#pragma unroll
    for (int r = 0; r < 8; ++r) {
      const int row = qtile * 16 + lg * 8 + r;   // C/D layout: M = r + 8*lg
      float s = acc[r] * 0.125f + mask[(size_t)row * SK_ + col];
      if (pad) s = NEG_INF;
      out[((size_t)bh * SQ_ + row) * (size_t)SK_ + col] = s;
    }
  }
}

extern "C" void kernel_launch(void* const* d_in, const int* in_sizes, int n_in,
                              void* d_out, int out_size, void* d_ws,
                              size_t ws_size, hipStream_t stream) {
  (void)in_sizes; (void)n_in; (void)out_size; (void)d_ws; (void)ws_size;
  const float* q    = (const float*)d_in[0];           // [B,H,SQ,D] f32
  const float* k    = (const float*)d_in[1];           // [B,H,SK,D] f32
  const float* mask = (const float*)d_in[2];           // [SQ,SK]    f32
  const unsigned char* kp = (const unsigned char*)d_in[3]; // [B,SK] bool
  float* out = (float*)d_out;                          // [B,H,SQ,SK] f32

  dim3 grid(SK_ / (128 * 8), SQ_ / 16, B_ * H_);       // (2, 128, 32)
  dim3 block(256);
  hipLaunchKernelGGL(sdp_scores_kernel, grid, block, 0, stream,
                     q, k, mask, kp, out);
}
